// GraphGPSModel_45389214384863
// MI455X (gfx1250) — matrix operand order repaired and checked by reading, not compiled, not run on hardware
//
#include <hip/hip_runtime.h>
#include <hip/hip_bf16.h>

// ---------------------------------------------------------------------------
// GraphGPS-like model: Linear -> TransformerConv(4 heads, mean) -> GCNConv -> Linear
// Dense GEMMs: v_wmma_f32_16x16x32_f16, wave32. Each wave owns a 16x64 output
// slab (4 accumulators, A fragment reused 4x). Weights staged transposed in
// LDS as f16 so each B fragment is one contiguous 32B LDS read (ds_load_b128).
// Edge attention / scatter via wave-per-edge kernels with L2-resident atomics.
// ---------------------------------------------------------------------------

#define NNODES 50000
#define NEDGES 800000
#define F_IN   128
#define HID    64
#define HEADS  4
#define QKV    (HEADS * HID)   // 256
#define F_OUT  64

typedef __attribute__((ext_vector_type(16))) _Float16 v16h;
typedef __attribute__((ext_vector_type(8)))  float    v8f;

// ------------------------- WMMA GEMM: C = A(MxK) @ W(KxN) + bias ------------
// Compile-time K (mult of 32) and NCOL (mult of 64). M % 16 == 0.
// Block = 256 threads = 8 waves; each wave -> one 16-row x 64-col slab.
template <int K, int NCOL>
__global__ __launch_bounds__(256)
void gemm_wmma(const float* __restrict__ A, const float* __restrict__ W,
               const float* __restrict__ bias, float* __restrict__ C,
               int M, int doRelu)
{
    // ---- stage W transposed in LDS as f16: Wt[n*K + k] ----
    __shared__ _Float16 Wt[NCOL * K];
    for (int idx = threadIdx.x; idx < K * NCOL; idx += 256) {
        const int kk = idx / NCOL;
        const int nn = idx - kk * NCOL;
        Wt[nn * K + kk] = (_Float16)W[idx];      // coalesced global read
    }
    __syncthreads();

    const int lane = threadIdx.x & 31;
    const int wave = threadIdx.x >> 5;
    constexpr int nColSlab = NCOL / 64;
    const int nRowT = M >> 4;
    const int slab  = blockIdx.x * 8 + wave;     // wave-uniform
    if (slab >= nRowT * nColSlab) return;        // after barrier; EXEC stays full
    const int rowT = slab / nColSlab;
    const int colS = slab - rowT * nColSlab;

    const int nloc = lane & 15;
    const int half = lane >> 4;
    const long rowA    = (long)(rowT * 16 + nloc);
    const int  colBase = colS * 64;

    v8f acc0 = {}, acc1 = {}, acc2 = {}, acc3 = {};
#pragma unroll
    for (int k0 = 0; k0 < K; k0 += 32) {
        // A 16x32 f16 fragment: lane holds one contiguous 16-float K run:
        //   [k0+half*16, +8) -> a[0..7],  [k0+half*16+8, +8) -> a[8..15]
        const float4* ap0 = (const float4*)(A + rowA * K + k0 + half * 16);
        const float4  f0 = ap0[0], f1 = ap0[1];
        const float4* ap1 = (const float4*)(A + rowA * K + k0 + half * 16 + 8);
        const float4  f2 = ap1[0], f3 = ap1[1];
        v16h a;
        a[0] = (_Float16)f0.x;  a[1] = (_Float16)f0.y;
        a[2] = (_Float16)f0.z;  a[3] = (_Float16)f0.w;
        a[4] = (_Float16)f1.x;  a[5] = (_Float16)f1.y;
        a[6] = (_Float16)f1.z;  a[7] = (_Float16)f1.w;
        a[8] = (_Float16)f2.x;  a[9] = (_Float16)f2.y;
        a[10] = (_Float16)f2.z; a[11] = (_Float16)f2.w;
        a[12] = (_Float16)f3.x; a[13] = (_Float16)f3.y;
        a[14] = (_Float16)f3.z; a[15] = (_Float16)f3.w;

        // B 32x16 fragment: lane needs W[k0+half*16 .. +15][col] ->
        // contiguous 16 f16 in transposed LDS (32B-aligned: 2x ds_load_b128)
        const v16h b0 = *(const v16h*)&Wt[(colBase + 0  + nloc) * K + k0 + half * 16];
        const v16h b1 = *(const v16h*)&Wt[(colBase + 16 + nloc) * K + k0 + half * 16];
        const v16h b2 = *(const v16h*)&Wt[(colBase + 32 + nloc) * K + k0 + half * 16];
        const v16h b3 = *(const v16h*)&Wt[(colBase + 48 + nloc) * K + k0 + half * 16];

        acc0 = __builtin_amdgcn_wmma_f32_16x16x32_f16(false, a, false, b0, (short)0, acc0, false, false);
        acc1 = __builtin_amdgcn_wmma_f32_16x16x32_f16(false, a, false, b1, (short)0, acc1, false, false);
        acc2 = __builtin_amdgcn_wmma_f32_16x16x32_f16(false, a, false, b2, (short)0, acc2, false, false);
        acc3 = __builtin_amdgcn_wmma_f32_16x16x32_f16(false, a, false, b3, (short)0, acc3, false, false);
    }

    // C/D layout: VGPR v -> row M = half*8 + v, column = lane&15 (+16 per frag)
    const v8f* accs[4] = { &acc0, &acc1, &acc2, &acc3 };
#pragma unroll
    for (int c = 0; c < 4; ++c) {
        const int ncol = colBase + c * 16 + nloc;
        const float bval = bias ? bias[ncol] : 0.0f;
#pragma unroll
        for (int vv = 0; vv < 8; ++vv) {
            const int mr = rowT * 16 + half * 8 + vv;
            float val = (*accs[c])[vv] + bval;
            if (doRelu) val = fmaxf(val, 0.0f);
            C[(long)mr * NCOL + ncol] = val;
        }
    }
}

// ------------------------- helpers ------------------------------------------
__device__ __forceinline__ void atomicMaxFloat(float* addr, float val)
{
    if (val >= 0.0f)
        atomicMax((int*)addr, __float_as_int(val));
    else
        atomicMin((unsigned int*)addr, __float_as_uint(val));
}

__global__ void fill_kernel(float* __restrict__ p, float val, long n)
{
    long i = (long)blockIdx.x * blockDim.x + threadIdx.x;
    if (i < n) p[i] = val;
}

// ------------------------- edge attention logits -----------------------------
// One wave per edge: alpha[e][h] = dot(q[dst][h], k[src][h]) / 8
__global__ __launch_bounds__(256)
void edge_alpha_kernel(const float* __restrict__ q, const float* __restrict__ k,
                       const int* __restrict__ src, const int* __restrict__ dst,
                       float* __restrict__ alpha, int E)
{
    const int wave = (int)(((long)blockIdx.x * blockDim.x + threadIdx.x) >> 5);
    const int lane = threadIdx.x & 31;
    if (wave >= E) return;
    const int s = src[wave], d = dst[wave];
    const int head = lane >> 3, sub = lane & 7;
    const float4* qp = (const float4*)(q + (long)d * QKV + head * HID + sub * 8);
    const float4* kp = (const float4*)(k + (long)s * QKV + head * HID + sub * 8);
    const float4 q0 = qp[0], q1 = qp[1];
    const float4 k0 = kp[0], k1 = kp[1];
    float sum = q0.x * k0.x + q0.y * k0.y + q0.z * k0.z + q0.w * k0.w
              + q1.x * k1.x + q1.y * k1.y + q1.z * k1.z + q1.w * k1.w;
    sum += __shfl_xor(sum, 1, 32);
    sum += __shfl_xor(sum, 2, 32);
    sum += __shfl_xor(sum, 4, 32);
    if (sub == 0) alpha[(long)wave * HEADS + head] = sum * 0.125f;  // 1/sqrt(64)
}

// ------------------------- segment max / exp / denom -------------------------
__global__ void seg_max_kernel(const float* __restrict__ alpha,
                               const int* __restrict__ dst,
                               float* __restrict__ m, int E)
{
    const long i = (long)blockIdx.x * blockDim.x + threadIdx.x;
    if (i >= (long)E * HEADS) return;
    const int e = (int)(i >> 2), h = (int)(i & 3);
    atomicMaxFloat(&m[(long)dst[e] * HEADS + h], alpha[i]);
}

__global__ void ea_denom_kernel(float* __restrict__ alpha,
                                const int* __restrict__ dst,
                                const float* __restrict__ m,
                                float* __restrict__ denom, int E)
{
    const long i = (long)blockIdx.x * blockDim.x + threadIdx.x;
    if (i >= (long)E * HEADS) return;
    const int e = (int)(i >> 2), h = (int)(i & 3);
    const float ea = __expf(alpha[i] - m[(long)dst[e] * HEADS + h]);
    alpha[i] = ea;
    atomicAdd(&denom[(long)dst[e] * HEADS + h], ea);
}

// ------------------------- weighted scatter of v -----------------------------
__global__ __launch_bounds__(256)
void agg_scatter_kernel(const float* __restrict__ ea, const float* __restrict__ denom,
                        const float* __restrict__ v, const int* __restrict__ src,
                        const int* __restrict__ dst, float* __restrict__ agg, int E)
{
    const int wave = (int)(((long)blockIdx.x * blockDim.x + threadIdx.x) >> 5);
    const int lane = threadIdx.x & 31;
    if (wave >= E) return;
    const int s = src[wave], d = dst[wave];
    const int head = lane >> 3, sub = lane & 7;
    const float w = ea[(long)wave * HEADS + head]
                  / (denom[(long)d * HEADS + head] + 1e-16f);
    const float* vp = v + (long)s * QKV + head * HID + sub * 8;
    float* op = agg + (long)d * QKV + head * HID + sub * 8;
#pragma unroll
    for (int c = 0; c < 8; ++c) atomicAdd(&op[c], w * vp[c]);
}

// ------------------------- head mean + skip + relu ---------------------------
__global__ void trans_finish_kernel(const float* __restrict__ agg,
                                    const float* __restrict__ skip,
                                    float* __restrict__ h2, int N)
{
    const long i = (long)blockIdx.x * blockDim.x + threadIdx.x;
    if (i >= (long)N * HID) return;
    const long n = i >> 6;
    const int  c = (int)(i & 63);
    const float* ap = agg + n * QKV + c;
    const float s = ap[0] + ap[64] + ap[128] + ap[192];
    h2[i] = fmaxf(0.25f * s + skip[i], 0.0f);
}

// ------------------------- GCN -----------------------------------------------
__global__ void deg_kernel(const int* __restrict__ dst, float* __restrict__ deg, int E)
{
    const long e = (long)blockIdx.x * blockDim.x + threadIdx.x;
    if (e < E) atomicAdd(&deg[dst[e]], 1.0f);
}

__global__ void dinv_kernel(float* __restrict__ deg, int N)
{
    const long i = (long)blockIdx.x * blockDim.x + threadIdx.x;
    if (i < N) deg[i] = rsqrtf(deg[i]);     // deg >= 1 (self loop)
}

__global__ void gcn_self_kernel(const float* __restrict__ xw,
                                const float* __restrict__ dinv,
                                float* __restrict__ acc, int N)
{
    const long i = (long)blockIdx.x * blockDim.x + threadIdx.x;
    if (i >= (long)N * HID) return;
    const float di = dinv[i >> 6];
    acc[i] = xw[i] * di * di;
}

__global__ __launch_bounds__(256)
void gcn_scatter_kernel(const float* __restrict__ xw, const float* __restrict__ dinv,
                        const int* __restrict__ src, const int* __restrict__ dst,
                        float* __restrict__ acc, int E)
{
    const int wave = (int)(((long)blockIdx.x * blockDim.x + threadIdx.x) >> 5);
    const int lane = threadIdx.x & 31;
    if (wave >= E) return;
    const int s = src[wave], d = dst[wave];
    const float norm = dinv[s] * dinv[d];
    const float2 xv = *((const float2*)(xw + (long)s * HID) + lane);
    float* op = acc + (long)d * HID + lane * 2;
    atomicAdd(&op[0], norm * xv.x);
    atomicAdd(&op[1], norm * xv.y);
}

__global__ void gcn_finish_kernel(const float* __restrict__ acc,
                                  const float* __restrict__ bg,
                                  float* __restrict__ h3, int N)
{
    const long i = (long)blockIdx.x * blockDim.x + threadIdx.x;
    if (i >= (long)N * HID) return;
    h3[i] = fmaxf(acc[i] + bg[i & 63], 0.0f);
}

// ---------------------------------------------------------------------------
extern "C" void kernel_launch(void* const* d_in, const int* in_sizes, int n_in,
                              void* d_out, int out_size, void* d_ws, size_t ws_size,
                              hipStream_t stream)
{
    const float* x     = (const float*)d_in[0];
    const int*   ei    = (const int*)  d_in[1];
    const float* W_in  = (const float*)d_in[2];
    const float* b_in  = (const float*)d_in[3];
    const float* Wq    = (const float*)d_in[4];
    const float* bq    = (const float*)d_in[5];
    const float* Wk    = (const float*)d_in[6];
    const float* bk    = (const float*)d_in[7];
    const float* Wv    = (const float*)d_in[8];
    const float* bv    = (const float*)d_in[9];
    const float* Wskip = (const float*)d_in[10];
    const float* bskip = (const float*)d_in[11];
    const float* Wg    = (const float*)d_in[12];
    const float* bg    = (const float*)d_in[13];
    const float* W_out = (const float*)d_in[14];
    const float* b_out = (const float*)d_in[15];

    const int* src = ei;
    const int* dst = ei + NEDGES;

    // -------- workspace layout (floats), with region reuse -----------------
    float* ws = (float*)d_ws;
    const long N64  = (long)NNODES * HID;    // 3.2 M
    const long N256 = (long)NNODES * QKV;    // 12.8 M
    const long E4   = (long)NEDGES * HEADS;  // 3.2 M
    const long N4   = (long)NNODES * HEADS;  // 0.2 M

    float* h1    = ws;                       // [N,64]
    float* q     = h1    + N64;              // [N,256]
    float* k     = q     + N256;             // [N,256]
    float* v     = k     + N256;             // [N,256]
    float* skip  = v     + N256;             // [N,64]
    float* alpha = skip  + N64;              // [E,4] (becomes ea in-place)
    float* mbuf  = alpha + E4;               // [N,4]
    float* denom = mbuf  + N4;               // [N,4]
    // reuse: q region free after edge_alpha; k region free after edge_alpha
    float* agg   = q;                        // [N,256] overlays q
    float* h2    = k;                        // [N,64]  overlays k
    float* xw    = k + N64;                  // [N,64]
    float* gcn   = k + 2 * N64;              // [N,64]
    float* deg   = k + 3 * N64;              // [N]
    float* h3    = h1;                       // [N,64]  overlays h1

    const int TPB = 256;
    auto blocks1d = [](long n, int tpb) { return (int)((n + tpb - 1) / tpb); };
    // slabs = (M/16 row tiles) * (NCOL/64 col slabs); 8 waves per block
    auto slabBlocks = [](int M, int Ncol) {
        return ((M >> 4) * (Ncol >> 6) + 7) / 8;
    };

    // -------- stage 0: input projection ------------------------------------
    gemm_wmma<F_IN, HID><<<slabBlocks(NNODES, HID), TPB, 0, stream>>>(
        x, W_in, b_in, h1, NNODES, 0);

    // -------- stage 1: q, k, v, skip projections ---------------------------
    gemm_wmma<HID, QKV><<<slabBlocks(NNODES, QKV), TPB, 0, stream>>>(
        h1, Wq, bq, q, NNODES, 0);
    gemm_wmma<HID, QKV><<<slabBlocks(NNODES, QKV), TPB, 0, stream>>>(
        h1, Wk, bk, k, NNODES, 0);
    gemm_wmma<HID, QKV><<<slabBlocks(NNODES, QKV), TPB, 0, stream>>>(
        h1, Wv, bv, v, NNODES, 0);
    gemm_wmma<HID, HID><<<slabBlocks(NNODES, HID), TPB, 0, stream>>>(
        h1, Wskip, bskip, skip, NNODES, 0);

    // -------- stage 2: edge attention --------------------------------------
    fill_kernel<<<blocks1d(N4, TPB), TPB, 0, stream>>>(mbuf, -__builtin_huge_valf(), N4);
    fill_kernel<<<blocks1d(N4, TPB), TPB, 0, stream>>>(denom, 0.0f, N4);

    edge_alpha_kernel<<<NEDGES / 8, TPB, 0, stream>>>(q, k, src, dst, alpha, NEDGES);
    seg_max_kernel<<<blocks1d(E4, TPB), TPB, 0, stream>>>(alpha, dst, mbuf, NEDGES);
    ea_denom_kernel<<<blocks1d(E4, TPB), TPB, 0, stream>>>(alpha, dst, mbuf, denom, NEDGES);

    // q/k regions now free: zero agg (overlays q) then scatter
    fill_kernel<<<blocks1d(N256, TPB), TPB, 0, stream>>>(agg, 0.0f, N256);
    agg_scatter_kernel<<<NEDGES / 8, TPB, 0, stream>>>(
        alpha, denom, v, src, dst, agg, NEDGES);
    trans_finish_kernel<<<blocks1d(N64, TPB), TPB, 0, stream>>>(agg, skip, h2, NNODES);

    // -------- stage 3: GCN --------------------------------------------------
    fill_kernel<<<blocks1d(NNODES, TPB), TPB, 0, stream>>>(deg, 1.0f, NNODES); // self loop
    deg_kernel<<<blocks1d(NEDGES, TPB), TPB, 0, stream>>>(dst, deg, NEDGES);
    dinv_kernel<<<blocks1d(NNODES, TPB), TPB, 0, stream>>>(deg, NNODES);

    gemm_wmma<HID, HID><<<slabBlocks(NNODES, HID), TPB, 0, stream>>>(
        h2, Wg, nullptr, xw, NNODES, 0);

    gcn_self_kernel<<<blocks1d(N64, TPB), TPB, 0, stream>>>(xw, deg, gcn, NNODES);
    gcn_scatter_kernel<<<NEDGES / 8, TPB, 0, stream>>>(xw, deg, src, dst, gcn, NEDGES);
    gcn_finish_kernel<<<blocks1d(N64, TPB), TPB, 0, stream>>>(gcn, bg, h3, NNODES);

    // -------- stage 4: output projection -----------------------------------
    gemm_wmma<HID, F_OUT><<<slabBlocks(NNODES, F_OUT), TPB, 0, stream>>>(
        h3, W_out, b_out, (float*)d_out, NNODES, 0);
}
